// DeepSSM_84404697301678
// MI455X (gfx1250) — matrix operand order repaired and checked by reading, not compiled, not run on hardware
//
#include <hip/hip_runtime.h>

typedef __attribute__((ext_vector_type(16))) __bf16 v16bf;
typedef __attribute__((ext_vector_type(8)))  float  v8f;

#define OBS_DIM 32
#define HIDDEN  64
#define GATES   256   // 4*HIDDEN
#define BATCH   256
#define TSTEPS  2048
#define ROWS    16    // batch rows per workgroup (WMMA M)

// sigmoid / tanh with v_exp_f32 + v_rcp_f32 (no IEEE division macro).
__device__ __forceinline__ float fast_sigmoid(float x) {
    return __builtin_amdgcn_rcpf(1.0f + __expf(-x));
}
__device__ __forceinline__ float fast_tanh(float x) {
    const float e = __expf(-2.0f * x);
    return (1.0f - e) * __builtin_amdgcn_rcpf(1.0f + e);
}

// Persistent batched-LSTM kernel.
// grid = BATCH/ROWS = 16 blocks of 256 threads (8 waves).
// Each block owns a 16-row batch tile; waves 0..7 own gate-column tiles
// {2w, 2w+1} (16 cols each). Weight B-fragments live in registers for the
// whole T loop; h/c state lives in LDS; h is re-deposited in bf16 A-fragment
// layout each step so the next step's WMMA reads it with 2x ds_load_b128.
// The x_{t+1} global load is software-pipelined over the element-wise phase.
__global__ __launch_bounds__(256) void
lstm_persistent(const float* __restrict__ y,     // [B, T, 32]
                const float* __restrict__ Wx,    // [32, 256]
                const float* __restrict__ Wh,    // [64, 256]
                const float* __restrict__ bias_g,// [256]
                float* __restrict__ out)         // [B, T, 64]
{
    __shared__ float zbuf[ROWS][GATES];               // 16 KB gate preacts
    __shared__ float cbuf[ROWS][HIDDEN];              // 4 KB cell state
    __shared__ __align__(32) __bf16 hfrag[2][32][16]; // 2 KB h in A-frag layout
    __shared__ __align__(32) __bf16 xfrag[32][16];    // 1 KB x in A-frag layout

    const int tid   = threadIdx.x;
    const int wave  = tid >> 5;
    const int lane  = tid & 31;
    const int b0    = blockIdx.x * ROWS;
    const int lncol = lane & 15;   // N column within 16x16 tile
    const int lkg   = lane >> 4;   // K/M group bit

    // ---------------- one-time: weight B-fragments -> registers -----------
    // 16-bit B (KxN=32x16): lane = ((k&15)>=8)*16 + n ; slot s = (k>=16?8:0) + (k&7).
    v16bf wxB[2];        // Wx: K=32, one fragment per ntile
    v16bf whB[2][2];     // Wh: K=64, two K-fragments per ntile
    float bval[2];
    for (int u = 0; u < 2; ++u) {
        const int n = (wave * 2 + u) * 16 + lncol;
        bval[u] = bias_g[n];
        for (int s = 0; s < 16; ++s) {
            const int k = (s & 8) * 2 + lkg * 8 + (s & 7);
            wxB[u][s] = (__bf16)Wx[k * GATES + n];
        }
        for (int j = 0; j < 2; ++j)
            for (int s = 0; s < 16; ++s) {
                const int k = j * 32 + (s & 8) * 2 + lkg * 8 + (s & 7);
                whB[u][j][s] = (__bf16)Wh[k * GATES + n];
            }
    }

    // ---------------- zero-init carry (c = 0, h = 0) -----------------------
    for (int i = tid; i < ROWS * HIDDEN; i += 256) ((float*)cbuf)[i] = 0.0f;
    for (int i = tid; i < 2 * 32 * 16; i += 256) ((__bf16*)hfrag)[i] = (__bf16)0.0f;

    // x-staging map for this thread: 16 rows x 16 float-pairs, coalesced.
    // 16-bit A (MxK=16x32): lane = (klo>=8)*16 + row ;
    //   vgpr v = (k>=16?4:0) + ((k&7)>>1), slot = 2v + (k&1).
    const int xr    = tid >> 4;
    const int xcp   = (tid & 15) * 2;
    const int xv    = (xcp < 16 ? 0 : 4) + ((xcp & 7) >> 1);
    const int xlane = (((xcp & 15) >= 8) ? 16 : 0) + xr;

    const float* xbase = y + (size_t)(b0 + xr) * TSTEPS * OBS_DIM + xcp;

    // prologue: fetch x_0 into registers (streamed once -> non-temporal)
    float xp0 = __builtin_nontemporal_load(xbase + 0);
    float xp1 = __builtin_nontemporal_load(xbase + 1);

    __syncthreads();

    for (int t = 0; t < TSTEPS; ++t) {
        // -------- commit prefetched x_t tile into bf16 A-fragment LDS
        xfrag[xlane][2 * xv]     = (__bf16)xp0;
        xfrag[xlane][2 * xv + 1] = (__bf16)xp1;
        __syncthreads();   // (A) x/h fragments visible; zbuf free to rewrite

        // -------- GEMM phase: z = x*Wx + h*Wh + b  (per wave: 6 WMMAs)
        v16bf xa  = *(const v16bf*)&xfrag[lane][0];
        v16bf ha0 = *(const v16bf*)&hfrag[0][lane][0];
        v16bf ha1 = *(const v16bf*)&hfrag[1][lane][0];

        for (int u = 0; u < 2; ++u) {
            v8f acc;
            for (int v = 0; v < 8; ++v) acc[v] = bval[u];   // bias pre-load
            acc = __builtin_amdgcn_wmma_f32_16x16x32_bf16(
                false, xa,  false, wxB[u],    (short)0, acc, false, false);
            acc = __builtin_amdgcn_wmma_f32_16x16x32_bf16(
                false, ha0, false, whB[u][0], (short)0, acc, false, false);
            acc = __builtin_amdgcn_wmma_f32_16x16x32_bf16(
                false, ha1, false, whB[u][1], (short)0, acc, false, false);
            // C/D layout: lane 0-15 -> M=v, lane 16-31 -> M=8+v ; N = lane&15
            const int ncol = (wave * 2 + u) * 16 + lncol;
            const int mrow = lkg * 8;
            for (int v = 0; v < 8; ++v) zbuf[mrow + v][ncol] = acc[v];
        }

        // -------- software-pipelined prefetch of x_{t+1} (clamped, no branch)
        {
            const int tn = (t + 1 < TSTEPS) ? (t + 1) : (TSTEPS - 1);
            const float* xp = xbase + (size_t)tn * OBS_DIM;
            xp0 = __builtin_nontemporal_load(xp + 0);
            xp1 = __builtin_nontemporal_load(xp + 1);
        }
        __syncthreads();   // (B) zbuf complete

        // -------- element-wise cell update: 16x64 = 4 elems / thread
        for (int e = 0; e < 4; ++e) {
            const int m  = e * 256 + tid;
            const int r  = m >> 6;
            const int hc = m & 63;
            const float ig = fast_sigmoid(zbuf[r][hc]);
            const float fg = fast_sigmoid(zbuf[r][64 + hc]);
            const float gg = fast_tanh(zbuf[r][128 + hc]);
            const float og = fast_sigmoid(zbuf[r][192 + hc]);
            const float c  = fg * cbuf[r][hc] + ig * gg;
            cbuf[r][hc] = c;
            const float h = og * fast_tanh(c);
            __builtin_nontemporal_store(
                h, &out[((size_t)(b0 + r) * TSTEPS + t) * HIDDEN + hc]);
            // deposit bf16 h into next step's A-fragment layout
            const int frag = hc >> 5;
            const int krem = hc & 31;
            const int v    = (krem < 16 ? 0 : 4) + ((krem & 7) >> 1);
            const int ln   = (((krem & 15) >= 8) ? 16 : 0) + r;
            hfrag[frag][ln][2 * v + (krem & 1)] = (__bf16)h;
        }
        // no barrier here: barrier (A) of the next iteration orders
        // hfrag writes vs. WMMA reads and zbuf reads vs. rewrites.
    }
}

extern "C" void kernel_launch(void* const* d_in, const int* in_sizes, int n_in,
                              void* d_out, int out_size, void* d_ws,
                              size_t ws_size, hipStream_t stream) {
    (void)in_sizes; (void)n_in; (void)d_ws; (void)ws_size; (void)out_size;
    const float* y  = (const float*)d_in[0];
    const float* Wx = (const float*)d_in[1];
    const float* Wh = (const float*)d_in[2];
    const float* b  = (const float*)d_in[3];
    float* out = (float*)d_out;
    lstm_persistent<<<BATCH / ROWS, 256, 0, stream>>>(y, Wx, Wh, b, out);
}
